// MambaModel_1116691497020
// MI455X (gfx1250) — matrix operand order repaired and checked by reading, not compiled, not run on hardware
//
#include <hip/hip_runtime.h>

// ---------------------------------------------------------------------------
// MI455X (gfx1250) Mamba forward.  wave32, WMMA bf16 GEMMs with async-to-LDS
// double-buffered B tiles (64-wide K chunks), scheduler-pipelined ds_load
// fragments, register-resident selective scan.
// ---------------------------------------------------------------------------

typedef __attribute__((ext_vector_type(16))) __bf16        v16bf;
typedef __attribute__((ext_vector_type(8)))  float         v8f;
typedef __attribute__((ext_vector_type(4)))  unsigned int  u32x4;

union FragBF { v16bf v; u32x4 q[2]; };

static __device__ __forceinline__ float siluf(float x) {
  return x / (1.0f + __expf(-x));
}
static __device__ __forceinline__ float softplusf(float x) {
  return (x > 20.0f) ? x : __logf(1.0f + __expf(x));
}

// Async DMA: global -> LDS, 16 bytes per lane.  ASYNCcnt-tracked.
static __device__ __forceinline__ void async_copy_b128(const void* g, void* l) {
  const unsigned int loff = (unsigned int)(uintptr_t)l;  // LDS offset = low 32 bits
  asm volatile("global_load_async_to_lds_b128 %0, %1, off"
               :: "v"(loff), "v"(g) : "memory");
}
static __device__ __forceinline__ void wait_async_4() {
  asm volatile("s_wait_asynccnt 0x4" ::: "memory");
}
static __device__ __forceinline__ void wait_async_0() {
  asm volatile("s_wait_asynccnt 0x0" ::: "memory");
}

// ---------------------------------------------------------------------------
// WMMA GEMM:  out = A[M,K](bf16,row) @ Wt[N,K](bf16, pre-transposed) + bias
// Block: 256 threads = 8 waves; wave w owns rows [m0+16w, +16), NT 16-col tiles.
// B tile staged in LDS by async DMA, double-buffered over 64-wide K chunks;
// sched_group_barrier pins DS reads one tile ahead of the consuming WMMA.
// EPI 0: silu + split cols into U / ZS (bf16);  1: softplus->f32;
// EPI 2: +residual->f32;  3: plain+bias->f32
// ---------------------------------------------------------------------------
template<int NT, int EPI>
__global__ __launch_bounds__(256)
void gemm_bf16_kernel(const __bf16* __restrict__ A, const __bf16* __restrict__ Wt,
                      const float* __restrict__ bias,
                      int M, int N, int K,
                      float*  __restrict__ outF,
                      __bf16* __restrict__ outU,
                      __bf16* __restrict__ outZ,
                      const float* __restrict__ res)
{
  constexpr int COLS   = NT * 16;   // columns per block
  constexpr int STRIDE = 144;       // bytes per column K-chunk (128 data + 16 pad)
  __shared__ __attribute__((aligned(16))) char sB[2][COLS * STRIDE];

  const int tid  = threadIdx.x;
  const int lane = tid & 31;
  const int wave = tid >> 5;
  const int m0   = blockIdx.x * 128 + wave * 16;
  const int n0   = blockIdx.y * COLS;
  const int hi   = lane >> 4;       // 0 or 1
  const int lo   = lane & 15;

  const __bf16* arow = A + (size_t)(m0 + lo) * K;

  // Cooperative async copy: unit = 64 bytes of one column's 128-byte K-chunk.
  const bool copier = tid < COLS * 2;
  const int  cc     = tid >> 1;     // column within tile
  const int  part   = tid & 1;      // which 64-byte half
  const __bf16* gcol = Wt + (size_t)(n0 + (copier ? cc : 0)) * K;

  auto issue = [&](int buf, int k0) {
    if (copier) {
      const char* g = (const char*)(gcol + k0) + part * 64;
      char* l = &sB[buf][cc * STRIDE + part * 64];
      async_copy_b128(g,      l);
      async_copy_b128(g + 16, l + 16);
      async_copy_b128(g + 32, l + 32);
      async_copy_b128(g + 48, l + 48);
    }
  };

  auto ldb = [&](FragBF& b, const char* base, int nt, int kk) {
    const char* bp = base + (size_t)(nt * 16 + lo) * STRIDE + kk * 64 + hi * 16;
    b.q[0] = *(const u32x4*)bp;
    b.q[1] = *(const u32x4*)(bp + 32);
  };

  v8f acc[NT] = {};
  const int nk = K / 64;

  issue(0, 0);                       // prologue: chunk 0 -> buffer 0

  for (int i = 0; i < nk; ++i) {
    const int k0 = i * 64;

    // Both A fragments for this chunk (global b128 x4), latency hidden
    // behind the async wait + barrier.
    FragBF a0, a1;
    a0.q[0] = *(const u32x4*)(arow + k0 + hi * 8);
    a0.q[1] = *(const u32x4*)(arow + k0 + 16 + hi * 8);
    a1.q[0] = *(const u32x4*)(arow + k0 + 32 + hi * 8);
    a1.q[1] = *(const u32x4*)(arow + k0 + 48 + hi * 8);
    if (i + 2 < nk)
      __builtin_prefetch(arow + k0 + 128, 0, 1);

    if (i + 1 < nk) {
      issue((i + 1) & 1, k0 + 64);
      wait_async_4();               // chunk i landed, chunk i+1 in flight
    } else {
      wait_async_0();
    }
    __syncthreads();                // publish buffer i&1 to all waves

    const char* base = sB[i & 1];
#pragma unroll
    for (int kk = 0; kk < 2; ++kk) {
      const FragBF& a = kk ? a1 : a0;
      FragBF bf[2];
      ldb(bf[0], base, 0, kk);      // pipelined: next tile's loads issued
#pragma unroll
      for (int nt = 0; nt < NT; ++nt) {          // before current is consumed
        if (nt + 1 < NT) ldb(bf[(nt + 1) & 1], base, nt + 1, kk);
        acc[nt] = __builtin_amdgcn_wmma_f32_16x16x32_bf16(
            false, a.v, false, bf[nt & 1].v, (short)0, acc[nt], false, false);
      }
    }

#if __has_builtin(__builtin_amdgcn_sched_group_barrier)
    // Pin the pipeline: 2 tiles of DS reads primed, then 1 WMMA : 2 DS reads,
    // ending with 2 WMMAs.  Per chunk: 4*NT ds_read, 2*NT wmma.
    __builtin_amdgcn_sched_group_barrier(0x100, 4, 0);   // DS read x4
#pragma unroll
    for (int u = 0; u < 2 * NT - 2; ++u) {
      __builtin_amdgcn_sched_group_barrier(0x008, 1, 0); // matrix op x1
      __builtin_amdgcn_sched_group_barrier(0x100, 2, 0); // DS read x2
    }
    __builtin_amdgcn_sched_group_barrier(0x008, 2, 0);   // matrix op x2
#endif

    __syncthreads();                // all waves done reading before overwrite
  }

#pragma unroll
  for (int nt = 0; nt < NT; ++nt) {
#pragma unroll
    for (int v = 0; v < 8; ++v) {
      const int row = m0 + v + 8 * hi;          // C/D layout: VGPR v, lane-half hi
      const int col = n0 + nt * 16 + lo;
      float val = acc[nt][v] + bias[col];
      if (EPI == 0) {
        const int half = N >> 1;
        const float s = siluf(val);
        if (col < half) outU[(size_t)row * half + col]          = (__bf16)s;
        else            outZ[(size_t)row * half + (col - half)] = (__bf16)s;
      } else if (EPI == 1) {
        outF[(size_t)row * N + col] = softplusf(val);
      } else if (EPI == 2) {
        outF[(size_t)row * N + col] = val + res[(size_t)row * N + col];
      } else {
        outF[(size_t)row * N + col] = val;
      }
    }
  }
}

// ---------------------------------------------------------------------------
// Selective scan. grid = (4, 16): blockIdx.y = batch, 256 ch/block.
// h[16] in VGPRs; B/C staged 32 timesteps at a time in LDS.
// ---------------------------------------------------------------------------
__global__ __launch_bounds__(256)
void scan_kernel(const __bf16* __restrict__ U, const float* __restrict__ DT,
                 const float* __restrict__ BC, const float* __restrict__ Alog,
                 const float* __restrict__ Dp, const __bf16* __restrict__ ZS,
                 __bf16* __restrict__ Y)
{
  const int d = blockIdx.x * 256 + threadIdx.x;   // 0..1023
  const int b = blockIdx.y;                       // 0..15
  __shared__ float sB[32][16];
  __shared__ float sC[32][16];

  float Av[16], h[16];
#pragma unroll
  for (int s = 0; s < 16; ++s) {
    Av[s] = -__expf(Alog[(size_t)d * 16 + s]);
    h[s]  = 0.0f;
  }
  const float Dd = Dp[d];
  const size_t base = (size_t)b * 2048;

  for (int tb = 0; tb < 2048; tb += 32) {
    __syncthreads();
#pragma unroll
    for (int i = 0; i < 4; ++i) {
      const int lin = threadIdx.x + i * 256;      // 32 steps * 32 vals
      const int tt = lin >> 5, s = lin & 31;
      const float v2 = BC[(base + tb + tt) * 32 + s];
      if (s < 16) sB[tt][s] = v2; else sC[tt][s - 16] = v2;
    }
    __syncthreads();
    for (int tt = 0; tt < 32; ++tt) {
      const size_t row = base + tb + tt;
      const float u   = (float)U[row * 1024 + d];
      const float dtv = DT[row * 1024 + d];
      const float dbu = dtv * u;
      float y = Dd * u;
#pragma unroll
      for (int s = 0; s < 16; ++s) {
        h[s] = h[s] * __expf(dtv * Av[s]) + dbu * sB[tt][s];
        y   += h[s] * sC[tt][s];
      }
      const float zs = (float)ZS[row * 1024 + d];
      Y[row * 1024 + d] = (__bf16)(y * zs);
    }
  }
}

// ---------------------------------------------------------------------------
// LayerNorm over rows of 512. One block per row.
// ---------------------------------------------------------------------------
__global__ __launch_bounds__(256)
void ln_kernel(const float* __restrict__ in, float* __restrict__ out,
               const float* __restrict__ g, const float* __restrict__ bb)
{
  const int row = blockIdx.x;
  const int tid = threadIdx.x;
  const float* p = in + (size_t)row * 512;
  __shared__ float red[256];

  const float v0 = p[tid], v1 = p[tid + 256];
  red[tid] = v0 + v1;
  __syncthreads();
  for (int o = 128; o > 0; o >>= 1) {
    if (tid < o) red[tid] += red[tid + o];
    __syncthreads();
  }
  const float mean = red[0] * (1.0f / 512.0f);
  __syncthreads();
  const float d0 = v0 - mean, d1 = v1 - mean;
  red[tid] = d0 * d0 + d1 * d1;
  __syncthreads();
  for (int o = 128; o > 0; o >>= 1) {
    if (tid < o) red[tid] += red[tid + o];
    __syncthreads();
  }
  const float r = rsqrtf(red[0] * (1.0f / 512.0f) + 1e-5f);
  float* q = out + (size_t)row * 512;
  q[tid]       = d0 * r * g[tid]       + bb[tid];
  q[tid + 256] = d1 * r * g[tid + 256] + bb[tid + 256];
}

// ---------------------------------------------------------------------------
// Embed: x[row, :512] = features[row, :32] @ embed_W + embed_b
// ---------------------------------------------------------------------------
__global__ __launch_bounds__(256)
void embed_kernel(const float* __restrict__ feat, const float* __restrict__ Wm,
                  const float* __restrict__ bm, float* __restrict__ x)
{
  const int row = blockIdx.x;
  __shared__ float f[32];
  if (threadIdx.x < 32) f[threadIdx.x] = feat[(size_t)row * 32 + threadIdx.x];
  __syncthreads();
  for (int c = threadIdx.x; c < 512; c += 256) {
    float acc = bm[c];
#pragma unroll
    for (int k = 0; k < 32; ++k) acc += f[k] * Wm[k * 512 + c];
    x[(size_t)row * 512 + c] = acc;
  }
}

// ---------------------------------------------------------------------------
// Weight transpose + f32->bf16:  Wt[n][k] = bf16(W[k][n])
// ---------------------------------------------------------------------------
__global__ __launch_bounds__(256)
void wtrans_kernel(const float* __restrict__ W, __bf16* __restrict__ Wt,
                   int K, int N)
{
  __shared__ __bf16 tile[32][33];
  const int tx = threadIdx.x & 31, ty = threadIdx.x >> 5; // 32x8
  const int nb = blockIdx.x * 32, kb = blockIdx.y * 32;
#pragma unroll
  for (int i = 0; i < 4; ++i) {
    const int k = kb + ty + i * 8;
    tile[ty + i * 8][tx] = (__bf16)W[(size_t)k * N + nb + tx];
  }
  __syncthreads();
#pragma unroll
  for (int i = 0; i < 4; ++i) {
    const int n = nb + ty + i * 8;
    Wt[(size_t)n * K + kb + tx] = tile[tx][ty + i * 8];
  }
}

__global__ __launch_bounds__(256)
void cvt_bf16_kernel(const float* __restrict__ in, __bf16* __restrict__ out, size_t n)
{
  const size_t i = (size_t)blockIdx.x * 256 + threadIdx.x;
  if (i < n) out[i] = (__bf16)in[i];
}

// ---------------------------------------------------------------------------
// Head: relu(x[:, -1, :] @ W1 + b1) @ W2 + b2 -> tanh.  One block per batch.
// ---------------------------------------------------------------------------
__global__ __launch_bounds__(256)
void head_kernel(const float* __restrict__ x, const float* __restrict__ W1,
                 const float* __restrict__ b1, const float* __restrict__ W2,
                 const float* __restrict__ b2, float* __restrict__ out)
{
  const int b = blockIdx.x;
  __shared__ float xp[512];
  __shared__ float red[256];
  const float* row = x + ((size_t)b * 2048 + 2047) * 512;
  xp[threadIdx.x]       = row[threadIdx.x];
  xp[threadIdx.x + 256] = row[threadIdx.x + 256];
  __syncthreads();
  const int j = threadIdx.x;
  float acc = b1[j];
  for (int k = 0; k < 512; ++k) acc += xp[k] * W1[(size_t)k * 256 + j];
  red[j] = fmaxf(acc, 0.0f) * W2[j];
  __syncthreads();
  for (int o = 128; o > 0; o >>= 1) {
    if (j < o) red[j] += red[j + o];
    __syncthreads();
  }
  if (j == 0) out[b] = tanhf(red[0] + b2[0]);
}

// ---------------------------------------------------------------------------
// Host orchestration
// ---------------------------------------------------------------------------
extern "C" void kernel_launch(void* const* d_in, const int* in_sizes, int n_in,
                              void* d_out, int out_size, void* d_ws, size_t ws_size,
                              hipStream_t stream)
{
  (void)in_sizes; (void)n_in; (void)out_size; (void)ws_size;

  const float* features = (const float*)d_in[1];
  const float* embed_W  = (const float*)d_in[2];
  const float* embed_b  = (const float*)d_in[3];
  const float* in_W     = (const float*)d_in[4];
  const float* in_b     = (const float*)d_in[5];
  const float* xp_W     = (const float*)d_in[6];
  const float* xp_b     = (const float*)d_in[7];
  const float* dt_W     = (const float*)d_in[8];
  const float* dt_b     = (const float*)d_in[9];
  const float* out_W    = (const float*)d_in[10];
  const float* out_b    = (const float*)d_in[11];
  const float* A_log    = (const float*)d_in[12];
  const float* D_param  = (const float*)d_in[13];
  const float* ln_g     = (const float*)d_in[14];
  const float* ln_b     = (const float*)d_in[15];
  const float* hW1      = (const float*)d_in[16];
  const float* hb1      = (const float*)d_in[17];
  const float* hW2      = (const float*)d_in[18];
  const float* hb2      = (const float*)d_in[19];

  const int ROWS = 16 * 2048;   // 32768 tokens

  size_t off = 0;
  auto wsalloc = [&](size_t bytes) -> void* {
    void* p = (char*)d_ws + off;
    off += (bytes + 255) & ~(size_t)255;
    return p;
  };

  float*  x     = (float*) wsalloc((size_t)ROWS * 512 * 4);
  float*  tmp   = (float*) wsalloc((size_t)ROWS * 512 * 4);
  __bf16* xb    = (__bf16*)wsalloc((size_t)ROWS * 512 * 2);
  __bf16* u_bf  = (__bf16*)wsalloc((size_t)ROWS * 1024 * 2);
  __bf16* zs_bf = (__bf16*)wsalloc((size_t)ROWS * 1024 * 2);
  __bf16* y_bf  = (__bf16*)wsalloc((size_t)ROWS * 1024 * 2);
  float*  dt_f  = (float*) wsalloc((size_t)ROWS * 1024 * 4);
  float*  bc_f  = (float*) wsalloc((size_t)ROWS * 32 * 4);
  __bf16* inWt  = (__bf16*)wsalloc((size_t)4 * 2048 * 512 * 2);
  __bf16* dtWt  = (__bf16*)wsalloc((size_t)4 * 1024 * 1024 * 2);
  __bf16* outWt = (__bf16*)wsalloc((size_t)4 * 512 * 1024 * 2);
  __bf16* xpWt  = (__bf16*)wsalloc((size_t)4 * 32 * 1024 * 2);

  // Weight transposes (f32 -> bf16, [K][N] -> [N][K]) for WMMA B operands.
  for (int l = 0; l < 4; ++l) {
    wtrans_kernel<<<dim3(2048/32, 512/32), 256, 0, stream>>>(
        in_W + (size_t)l * 512 * 2048, inWt + (size_t)l * 2048 * 512, 512, 2048);
    wtrans_kernel<<<dim3(1024/32, 1024/32), 256, 0, stream>>>(
        dt_W + (size_t)l * 1024 * 1024, dtWt + (size_t)l * 1024 * 1024, 1024, 1024);
    wtrans_kernel<<<dim3(512/32, 1024/32), 256, 0, stream>>>(
        out_W + (size_t)l * 1024 * 512, outWt + (size_t)l * 512 * 1024, 1024, 512);
    wtrans_kernel<<<dim3(32/32, 1024/32), 256, 0, stream>>>(
        xp_W + (size_t)l * 1024 * 32, xpWt + (size_t)l * 32 * 1024, 1024, 32);
  }

  embed_kernel<<<ROWS, 256, 0, stream>>>(features, embed_W, embed_b, x);

  for (int l = 0; l < 4; ++l) {
    const size_t nx = (size_t)ROWS * 512;
    cvt_bf16_kernel<<<(unsigned)((nx + 255) / 256), 256, 0, stream>>>(x, xb, nx);

    // in-proj: fused SiLU, split into u / silu(z)
    gemm_bf16_kernel<8, 0><<<dim3(256, 16), 256, 0, stream>>>(
        xb, inWt + (size_t)l * 2048 * 512, in_b + (size_t)l * 2048,
        ROWS, 2048, 512, nullptr, u_bf, zs_bf, nullptr);

    // xp-proj -> B,C  (N=32)
    gemm_bf16_kernel<2, 3><<<dim3(256, 1), 256, 0, stream>>>(
        u_bf, xpWt + (size_t)l * 32 * 1024, xp_b + (size_t)l * 32,
        ROWS, 32, 1024, bc_f, nullptr, nullptr, nullptr);

    // dt-proj: fused softplus -> f32
    gemm_bf16_kernel<8, 1><<<dim3(256, 8), 256, 0, stream>>>(
        u_bf, dtWt + (size_t)l * 1024 * 1024, dt_b + (size_t)l * 1024,
        ROWS, 1024, 1024, dt_f, nullptr, nullptr, nullptr);

    // selective scan, fused * silu(z) -> bf16 y
    scan_kernel<<<dim3(4, 16), 256, 0, stream>>>(
        u_bf, dt_f, bc_f, A_log + (size_t)l * 1024 * 16,
        D_param + (size_t)l * 1024, zs_bf, y_bf);

    // out-proj: fused bias + residual add -> tmp
    gemm_bf16_kernel<8, 2><<<dim3(256, 4), 256, 0, stream>>>(
        y_bf, outWt + (size_t)l * 512 * 1024, out_b + (size_t)l * 512,
        ROWS, 512, 1024, tmp, nullptr, nullptr, x);

    ln_kernel<<<ROWS, 256, 0, stream>>>(tmp, x, ln_g + (size_t)l * 512,
                                        ln_b + (size_t)l * 512);
  }

  head_kernel<<<16, 256, 0, stream>>>(x, hW1, hb1, hW2, hb2, (float*)d_out);
}
